// SkipableGAT_6236292514448
// MI455X (gfx1250) — matrix operand order repaired and checked by reading, not compiled, not run on hardware
//
#include <hip/hip_runtime.h>
#include <hip/hip_bf16.h>
#include <math.h>

typedef __attribute__((ext_vector_type(16))) _Float16 v16h;
typedef __attribute__((ext_vector_type(8)))  _Float16 v8h;
typedef __attribute__((ext_vector_type(8)))  float    v8f;

#define HNUM   8
#define JC     24          // padded joints (Cayley graph size)
#define JS     17          // skeleton joints
#define CDIM   256
#define ODIM   1536        // 6*C (qkv output)
#define BTOT   1296        // B*T = 16*81
#define EMAX   120
#define BETA_  0.8f
#define CAT5   1280        // 5*C

// dynamic LDS budget for the fused layer kernel (CDNA5: 320KB/WGP available)
//  qkv f32 24*1536       = 147456 B
//  agg f32 24*8*32       =  24576 B
//  s   f32 120*8         =   3840 B
//  sig f32 24*8          =    768 B
//  hmax f32 8            =     32 B
//  aw  f32 64            =    256 B
//  src/dst int 2*120     =    960 B
//  A   f16 32*256        =  16384 B
#define LAYER_SMEM 194304

__device__ __forceinline__ float softplusf_(float x) {
  // matches jax.nn.softplus: max(x,0) + log1p(exp(-|x|))
  return fmaxf(x, 0.f) + log1pf(expf(-fabsf(x)));
}

// ---------------------------------------------------------------- pad / seed
extern "C" __global__ void k_pad(const float* __restrict__ x,
                                 float* __restrict__ xc32,
                                 _Float16* __restrict__ xc16,
                                 _Float16* __restrict__ ycat)
{
  int t = blockIdx.x * 256 + threadIdx.x;          // over BTOT*JC*CDIM
  if (t >= BTOT * JC * CDIM) return;
  int c  = t & 255;
  int r  = t >> 8;
  int j  = r % JC;
  int bt = r / JC;
  float v = 0.f;
  if (j < JS) v = x[(bt * JS + j) * CDIM + c];
  xc32[t] = v;
  xc16[t] = (_Float16)v;
  if (j < JS) ycat[(size_t)(bt * JS + j) * CAT5 + c] = (_Float16)v;  // slot 0
}

// ---------------------------------------------------------------- f32 -> f16
extern "C" __global__ void k_cvt(const float* __restrict__ s,
                                 _Float16* __restrict__ d, int n)
{
  int t = blockIdx.x * 256 + threadIdx.x;
  if (t < n) d[t] = (_Float16)s[t];
}

// ------------------------------------------------- fused GAT layer (1 bt/blk)
extern "C" __global__ __launch_bounds__(256)
void k_layer(float* __restrict__ xc32, _Float16* __restrict__ xc16,
             const _Float16* __restrict__ w16, const float* __restrict__ aw,
             const float* __restrict__ gamma, const float* __restrict__ beta,
             const int* __restrict__ edges, int E,
             _Float16* __restrict__ ycat, int slot)
{
  extern __shared__ char smem[];
  float*    qkv_l  = (float*)smem;                   // [24][1536]
  float*    agg_l  = qkv_l + JC * ODIM;              // [24][256]
  float*    s_l    = agg_l + JC * CDIM;              // [120][8]
  float*    sig_l  = s_l + EMAX * HNUM;              // [24][8]
  float*    hmax_l = sig_l + JC * HNUM;              // [8]
  float*    aw_l   = hmax_l + HNUM;                  // [64]
  int*      src_l  = (int*)(aw_l + 64);              // [120]
  int*      dst_l  = src_l + EMAX;                   // [120]
  _Float16* A_l    = (_Float16*)(dst_l + EMAX);      // [32][256], 16B aligned

  const int tid  = threadIdx.x;
  const int bt   = blockIdx.x;
  const int wave = tid >> 5;
  const int lane = tid & 31;
  const int lh   = lane & 15;         // lane % 16
  const int hi   = lane >> 4;         // lane-half select

  // ---- stage edges, a-vector, activations (f16, rows >=24 zero) ----
  for (int t = tid; t < E; t += 256) { src_l[t] = edges[t]; dst_l[t] = edges[E + t]; }
  for (int t = tid; t < 64; t += 256) aw_l[t] = aw[t];
  for (int t = tid; t < 32 * CDIM; t += 256) {
    int j = t >> 8;
    A_l[t] = (j < JC) ? xc16[(size_t)(bt * JC + j) * CDIM + (t & 255)] : (_Float16)0.f;
  }
  __syncthreads();

  // ---- QKV GEMM: [32 x 256] x [256 x 1536] via v_wmma_f32_16x16x32_f16 ----
  // Each wave owns an N-tile column and both M-tiles (shares B-fragment).
  for (int nt = wave; nt < ODIM / 16; nt += 8) {
    v8f acc0 = {}; v8f acc1 = {};
    const _Float16* wp = w16 + (size_t)(nt * 16 + lh) * CDIM + (hi ? 16 : 0);
    for (int kb = 0; kb < CDIM / 32; ++kb) {
      // B fragment: lanes 0-15 K=0..15 col=lane; lanes 16-31 K=16..31
      v16h bf = *(const v16h*)(wp + kb * 32);
      // A fragments (contiguous K order, matches B): two contiguous 8-half chunks
      int ab = kb * 32 + (hi ? 16 : 0);
      v8h a0lo = *(const v8h*)(A_l + lh * CDIM + ab);
      v8h a0hi = *(const v8h*)(A_l + lh * CDIM + ab + 8);
      v8h a1lo = *(const v8h*)(A_l + (16 + lh) * CDIM + ab);
      v8h a1hi = *(const v8h*)(A_l + (16 + lh) * CDIM + ab + 8);
      v16h af0, af1;
      for (int i = 0; i < 8; ++i) {
        af0[i] = a0lo[i]; af0[i + 8] = a0hi[i];
        af1[i] = a1lo[i]; af1[i + 8] = a1hi[i];
      }
      acc0 = __builtin_amdgcn_wmma_f32_16x16x32_f16(false, af0, false, bf,
                                                    (short)0, acc0, false, false);
      acc1 = __builtin_amdgcn_wmma_f32_16x16x32_f16(false, af1, false, bf,
                                                    (short)0, acc1, false, false);
    }
    int col = nt * 16 + lh;
    for (int r = 0; r < 8; ++r) {
      int row0 = r + (hi ? 8 : 0);
      qkv_l[row0 * ODIM + col] = acc0[r];
      int row1 = 16 + row0;
      if (row1 < JC) qkv_l[row1 * ODIM + col] = acc1[r];
    }
  }
  __syncthreads();

  // ---- per-edge logits: s = sum_a aw[a]*softplus(q[src]+k[dst]) ----
  const int EH = E * HNUM;
  for (int t = tid; t < EH; t += 256) {
    int e = t >> 3, h = t & 7;
    const float* qp = qkv_l + src_l[e] * ODIM + h * 192;
    const float* kp = qkv_l + dst_l[e] * ODIM + h * 192 + 64;
    float s = 0.f;
    for (int a = 0; a < 64; ++a) s += softplusf_(qp[a] + kp[a]) * aw_l[a];
    s_l[t] = s;
  }
  __syncthreads();
  if (tid < HNUM) {                       // global max over edges per head
    float m = -3.0e38f;
    for (int e = 0; e < E; ++e) m = fmaxf(m, s_l[e * HNUM + tid]);
    hmax_l[tid] = m;
  }
  __syncthreads();
  for (int t = tid; t < EH; t += 256) s_l[t] = expf(s_l[t] - hmax_l[t & 7]);
  __syncthreads();
  // deterministic segment sums (no float atomics -> bit-stable replays)
  for (int t = tid; t < JC * HNUM; t += 256) {
    int j = t >> 3, h = t & 7;
    float sum = 1e-10f;
    for (int e = 0; e < E; ++e) if (src_l[e] == j) sum += s_l[e * HNUM + h];
    sig_l[t] = sum;
  }
  __syncthreads();
  for (int t = tid; t < EH; t += 256)
    s_l[t] /= sig_l[src_l[t >> 3] * HNUM + (t & 7)];
  __syncthreads();
  // ---- aggregate messages: agg[j,h,d] = sum_{e:src==j} attn * v2[dst] ----
  for (int t = tid; t < JC * CDIM; t += 256) {
    int j = t >> 8, c = t & 255, h = c >> 5, d = c & 31;
    float acc = 0.f;
    for (int e = 0; e < E; ++e)
      if (src_l[e] == j)
        acc += s_l[e * HNUM + h] * qkv_l[dst_l[e] * ODIM + h * 192 + 160 + d];
    agg_l[t] = acc;
  }
  __syncthreads();

  // ---- residual + LayerNorm, write xc (f32+f16) and ycat slot ----
  for (int j = wave; j < JC; j += 8) {
    float vals[8], sum = 0.f, sq = 0.f;
    for (int i = 0; i < 8; ++i) {
      int c = lane + 32 * i;              // h = i, d = lane
      float v1 = qkv_l[j * ODIM + i * 192 + 128 + lane];
      float v  = xc32[(size_t)(bt * JC + j) * CDIM + c]
               + (1.f - BETA_) * v1 + BETA_ * agg_l[j * CDIM + c];
      vals[i] = v; sum += v; sq += v * v;
    }
    for (int m = 16; m >= 1; m >>= 1) {
      sum += __shfl_xor(sum, m, 32);
      sq  += __shfl_xor(sq,  m, 32);
    }
    float mean = sum * (1.f / 256.f);
    float var  = sq * (1.f / 256.f) - mean * mean;
    float rstd = rsqrtf(var + 1e-5f);
    for (int i = 0; i < 8; ++i) {
      int c = lane + 32 * i;
      float o = (vals[i] - mean) * rstd * gamma[c] + beta[c];
      size_t idx = (size_t)(bt * JC + j) * CDIM + c;
      xc32[idx] = o;
      xc16[idx] = (_Float16)o;
      if (j < JS)
        ycat[(size_t)(bt * JS + j) * CAT5 + slot * CDIM + c] = (_Float16)o;
    }
  }
}

// ------------------------------------------- final projection GEMM (+ bias)
extern "C" __global__ __launch_bounds__(256)
void k_proj(const _Float16* __restrict__ ycat, const _Float16* __restrict__ pw,
            const float* __restrict__ pb, float* __restrict__ out)
{
  int rt   = blockIdx.x;                 // 0..1376 row tiles (22032/16)
  int wave = threadIdx.x >> 5;
  int lane = threadIdx.x & 31;
  int lh   = lane & 15, hi = lane >> 4;
  int nt   = blockIdx.y * 8 + wave;      // 0..15 col tiles
  v8f acc = {};
  const _Float16* ap = ycat + (size_t)(rt * 16 + lh) * CAT5 + (hi ? 16 : 0);
  const _Float16* bp = pw   + (size_t)(nt * 16 + lh) * CAT5 + (hi ? 16 : 0);
  for (int kb = 0; kb < CAT5 / 32; ++kb) {
    v8h alo = *(const v8h*)(ap + kb * 32);
    v8h ahi = *(const v8h*)(ap + kb * 32 + 8);
    v16h af;
    for (int i = 0; i < 8; ++i) { af[i] = alo[i]; af[i + 8] = ahi[i]; }
    v16h bf = *(const v16h*)(bp + kb * 32);
    acc = __builtin_amdgcn_wmma_f32_16x16x32_f16(false, af, false, bf,
                                                 (short)0, acc, false, false);
  }
  int col = nt * 16 + lh;
  float bias = pb[col];
  for (int r = 0; r < 8; ++r) {
    int row = rt * 16 + r + (hi ? 8 : 0);
    out[(size_t)row * CDIM + col] = acc[r] + bias;
  }
}

// ------------------------------------------------------- final LayerNorm
extern "C" __global__ __launch_bounds__(256)
void k_lnout(const float* __restrict__ in, const float* __restrict__ g,
             const float* __restrict__ b, float* __restrict__ out)
{
  int row  = blockIdx.x * 8 + (threadIdx.x >> 5);   // one wave per row
  int lane = threadIdx.x & 31;
  const float* p = in + (size_t)row * CDIM;
  float vals[8], sum = 0.f, sq = 0.f;
  for (int i = 0; i < 8; ++i) {
    float v = p[lane + 32 * i]; vals[i] = v; sum += v; sq += v * v;
  }
  for (int m = 16; m >= 1; m >>= 1) {
    sum += __shfl_xor(sum, m, 32);
    sq  += __shfl_xor(sq,  m, 32);
  }
  float mean = sum * (1.f / 256.f);
  float var  = sq * (1.f / 256.f) - mean * mean;
  float rstd = rsqrtf(var + 1e-5f);
  float* o = out + (size_t)row * CDIM;
  for (int i = 0; i < 8; ++i) {
    int c = lane + 32 * i;
    o[c] = (vals[i] - mean) * rstd * g[c] + b[c];
  }
}

// ---------------------------------------------------------------- launcher
extern "C" void kernel_launch(void* const* d_in, const int* in_sizes, int n_in,
                              void* d_out, int out_size, void* d_ws, size_t ws_size,
                              hipStream_t stream)
{
  (void)n_in; (void)out_size; (void)ws_size;
  const float* x  = (const float*)d_in[0];
  const float* wqkv[4] = {(const float*)d_in[1], (const float*)d_in[5],
                          (const float*)d_in[9], (const float*)d_in[13]};
  const float* aw[4]   = {(const float*)d_in[2], (const float*)d_in[6],
                          (const float*)d_in[10], (const float*)d_in[14]};
  const float* gg[4]   = {(const float*)d_in[3], (const float*)d_in[7],
                          (const float*)d_in[11], (const float*)d_in[15]};
  const float* bb[4]   = {(const float*)d_in[4], (const float*)d_in[8],
                          (const float*)d_in[12], (const float*)d_in[16]};
  const float* pw = (const float*)d_in[17];
  const float* pb = (const float*)d_in[18];
  const float* gp = (const float*)d_in[19];
  const float* bp = (const float*)d_in[20];
  const int* e_skel = (const int*)d_in[21];
  const int* e_cay  = (const int*)d_in[22];
  const int E_skel = in_sizes[21] / 2;   // 49
  const int E_cay  = in_sizes[22] / 2;   // 120

  // workspace carving (256B aligned)
  char* ws = (char*)d_ws;
  size_t off = 0;
  auto carve = [&](size_t bytes) -> void* {
    void* p = ws + off;
    off += (bytes + 255) & ~(size_t)255;
    return p;
  };
  float*    xc32  = (float*)   carve((size_t)BTOT * JC * CDIM * 4);
  _Float16* xc16  = (_Float16*)carve((size_t)BTOT * JC * CDIM * 2);
  _Float16* wf16  = (_Float16*)carve((size_t)4 * ODIM * CDIM * 2);
  _Float16* pwf16 = (_Float16*)carve((size_t)CDIM * CAT5 * 2);
  _Float16* ycat  = (_Float16*)carve((size_t)BTOT * JS * CAT5 * 2);
  float*    ptmp  = (float*)   carve((size_t)BTOT * JS * CDIM * 4);

  // 1) pad x -> xc (f32+f16), seed ycat slot 0
  k_pad<<<BTOT * JC, 256, 0, stream>>>(x, xc32, xc16, ycat);

  // 2) convert weights to f16 (compute path the chip rewards)
  for (int l = 0; l < 4; ++l)
    k_cvt<<<(ODIM * CDIM + 255) / 256, 256, 0, stream>>>(
        wqkv[l], wf16 + (size_t)l * ODIM * CDIM, ODIM * CDIM);
  k_cvt<<<(CDIM * CAT5 + 255) / 256, 256, 0, stream>>>(pw, pwf16, CDIM * CAT5);

  // 3) four fused GAT layers (WMMA GEMM + edge softmax + residual + LN)
  const int* elist[4] = {e_skel, e_cay, e_skel, e_cay};
  const int  ecnt[4]  = {E_skel, E_cay, E_skel, E_cay};
  for (int l = 0; l < 4; ++l)
    k_layer<<<BTOT, 256, LAYER_SMEM, stream>>>(
        xc32, xc16, wf16 + (size_t)l * ODIM * CDIM,
        aw[l], gg[l], bb[l], elist[l], ecnt[l], ycat, l + 1);

  // 4) projection GEMM (22032x1280 @ 1280x256) + bias, then LayerNorm
  k_proj<<<dim3((BTOT * JS) / 16, 2), 256, 0, stream>>>(ycat, pwf16, pb, ptmp);
  k_lnout<<<(BTOT * JS) / 8, 256, 0, stream>>>(ptmp, gp, bp, (float*)d_out);
}
